// MultiHeadedAttention_35424890257945
// MI455X (gfx1250) — matrix operand order repaired and checked by reading, not compiled, not run on hardware
//
#include <hip/hip_runtime.h>
#include <hip/hip_bf16.h>

// MI455X / gfx1250, wave32. All GEMMs on v_wmma_f32_16x16x32_bf16.
// B=1, T=4096, D=1024, H=16, DK=DV=64, L=32, ROT_SCALE=1 (and 1/ROT_SCALE=1).

#define T_SEQ   4096
#define D_MODEL 1024
#define NH      16
#define HDK     64
#define HDV     64
#define LROT    32

typedef __attribute__((ext_vector_type(16))) __bf16 v16bf;
typedef __attribute__((ext_vector_type(8)))  float  v8f;
typedef __attribute__((ext_vector_type(4)))  int    v4i;

typedef __attribute__((address_space(1))) v4i* gv4i_p;   // global int4*
typedef __attribute__((address_space(3))) v4i* lv4i_p;   // LDS int4*

union FragU { v16bf v; uint4 q[2]; };

#if __has_builtin(__builtin_amdgcn_global_load_async_to_lds_b128)
#define HAVE_ASYNC_LDS 1
#endif

#define WMMA_BF16(A, B, C) __builtin_amdgcn_wmma_f32_16x16x32_bf16( \
    false, (A), false, (B), (short)0, (C), false, false)

// ---- helpers ------------------------------------------------------------
__device__ inline unsigned pack2bf(float a, float b) {
  union { __bf16 h[2]; unsigned u; } r;
  r.h[0] = (__bf16)a; r.h[1] = (__bf16)b;   // -> v_cvt_pk_bf16_f32
  return r.u;
}
// store 4 contiguous bf16 (from float4) as one 8B LDS store
__device__ inline void store4bf(__bf16* dst, float4 v) {
  uint2 p; p.x = pack2bf(v.x, v.y); p.y = pack2bf(v.z, v.w);
  *(uint2*)dst = p;
}
// 16B global -> LDS copy; async path (ASYNCcnt) on toolchains that expose it
__device__ inline void copy16_g2l(__bf16* lds_dst, const __bf16* gsrc) {
#ifdef HAVE_ASYNC_LDS
  const void* gv = (const void*)gsrc;
  void* lv = (void*)lds_dst;
  __builtin_amdgcn_global_load_async_to_lds_b128((gv4i_p)gv, (lv4i_p)lv, 0, 0);
#else
  *(uint4*)lds_dst = *(const uint4*)gsrc;
#endif
}
__device__ inline void wait_async_done() {
#ifdef HAVE_ASYNC_LDS
#if __has_builtin(__builtin_amdgcn_s_wait_asynccnt)
  __builtin_amdgcn_s_wait_asynccnt(0);
#else
  asm volatile("s_wait_asynccnt 0" ::: "memory");
#endif
#endif
}

// ---- CDNA5 WMMA fragment loaders (16x16x32 bf16) ------------------------
// A-matrix 16x32 (MxK): lane L holds row M=L%16; lanes 0-15 hold K=[0..7]+[16..23],
// lanes 16-31 hold K=[8..15]+[24..31]  (ISA 7.12.2, 16-bit A 16x32 table).
__device__ inline v16bf load_frag_a(const __bf16* base, int stride, int row,
                                    int k0, int hi) {
  FragU f;
  const __bf16* p = base + (size_t)row * stride + k0 + 8 * hi;
  f.q[0] = *(const uint4*)(p);
  f.q[1] = *(const uint4*)(p + 16);
  return f.v;
}
// B-matrix 32x16 (KxN): lane L holds column N=L%16; lanes 0-15 hold K=[0..7]+[16..23],
// lanes 16-31 hold K=[8..15]+[24..31]. "base" is stored [N][K] row-major, so per-lane
// data (one column of Bwmma = one stored row) is contiguous.
__device__ inline v16bf load_frag_b(const __bf16* base, int stride, int row,
                                    int k0, int hi) {
  FragU f;
  const __bf16* p = base + (size_t)row * stride + k0 + 8 * hi;
  f.q[0] = *(const uint4*)(p);
  f.q[1] = *(const uint4*)(p + 16);
  return f.v;
}

// ---- Kernel 1: projection GEMM + fused RoPE epilogue --------------------
// Block: 256 rows x one head (64 cols). Wave: 32 rows x 64 cols (8 C-tiles).
// Y[t, h*64+c] = sum_d X[t,d] * W[h*64+c, d], then RoPE on c<32.
// transpose_out==0: Out layout [H][T][64] (bf16).  ==1: [H][64][T] (V-transposed).
__global__ __launch_bounds__(256) void proj_rope_kernel(
    const float* __restrict__ X, const float* __restrict__ W,
    const float* __restrict__ freqs, __bf16* __restrict__ Out,
    float scale, int transpose_out) {
  __shared__ __bf16 Xs[256][72];  // k-depth 64, +8 pad (row stride 144B, 16B-aligned)
  __shared__ __bf16 Ws[64][72];

  const int tid  = threadIdx.x;
  const int wv   = tid >> 5;
  const int lane = tid & 31;
  const int hi   = lane >> 4;
  const int ln   = lane & 15;
  const int rowBase = blockIdx.x * 256;
  const int h       = blockIdx.y;

  v8f acc[2][4] = {};

  for (int d0 = 0; d0 < D_MODEL; d0 += 64) {
    __syncthreads();
#pragma unroll
    for (int i = 0; i < 16; ++i) {            // 256x64 fp32 -> bf16, float4 loads
      int idx = tid + i * 256;
      int r = idx >> 4, c4 = (idx & 15) * 4;
      float4 xf = *(const float4*)&X[(size_t)(rowBase + r) * D_MODEL + d0 + c4];
      store4bf(&Xs[r][c4], xf);
    }
#pragma unroll
    for (int i = 0; i < 4; ++i) {             // 64x64 weight tile for head h
      int idx = tid + i * 256;
      int r = idx >> 4, c4 = (idx & 15) * 4;
      float4 wf = *(const float4*)&W[(size_t)(h * HDK + r) * D_MODEL + d0 + c4];
      store4bf(&Ws[r][c4], wf);
    }
    __syncthreads();

    // Issue ALL fragment loads first, then the 16 WMMAs (DS/WMMA overlap).
    v16bf a[2][2], b[4][2];
#pragma unroll
    for (int rt = 0; rt < 2; ++rt)
#pragma unroll
      for (int dk = 0; dk < 2; ++dk)
        a[rt][dk] = load_frag_a(&Xs[0][0], 72, wv * 32 + rt * 16 + ln, dk * 32, hi);
#pragma unroll
    for (int t = 0; t < 4; ++t)
#pragma unroll
      for (int dk = 0; dk < 2; ++dk)
        b[t][dk] = load_frag_b(&Ws[0][0], 72, t * 16 + ln, dk * 32, hi);
#pragma unroll
    for (int rt = 0; rt < 2; ++rt)
#pragma unroll
      for (int t = 0; t < 4; ++t)
#pragma unroll
        for (int dk = 0; dk < 2; ++dk)
          acc[rt][t] = WMMA_BF16(a[rt][dk], b[t][dk], acc[rt][t]);
  }

  // RoPE on cols [0,32): pairs (c, c+16) live in acc[rt][0]/acc[rt][1] same reg+lane.
#pragma unroll
  for (int rt = 0; rt < 2; ++rt) {
    const int myRow0 = rowBase + wv * 32 + rt * 16;
#pragma unroll
    for (int r = 0; r < 8; ++r) {
      int t = myRow0 + r + 8 * hi;
      float f0 = freqs[t * LROT + ln];
      float f1 = freqs[t * LROT + 16 + ln];
      float a0 = acc[rt][0][r], a1 = acc[rt][1][r];
      acc[rt][0][r] = (a0 * __cosf(f0) - a1 * __sinf(f0)) * scale;
      acc[rt][1][r] = (a1 * __cosf(f1) + a0 * __sinf(f1)) * scale;
    }
#pragma unroll
    for (int tt = 0; tt < 4; ++tt) {
#pragma unroll
      for (int r = 0; r < 8; ++r) {
        int t = myRow0 + r + 8 * hi;
        int c = tt * 16 + ln;
        __bf16 val = (__bf16)acc[rt][tt][r];
        if (transpose_out)
          Out[(size_t)h * HDV * T_SEQ + (size_t)c * T_SEQ + t] = val;
        else
          Out[((size_t)h * T_SEQ + t) * HDK + c] = val;
      }
    }
  }
}

// ---- Kernel 2: causal flash attention (bf16 WMMA, f32 online softmax) ---
// Qh,Kh: [H][T][64] bf16. VhT: [H][64][T] bf16. AttnOut: [T][H*64] bf16.
__global__ __launch_bounds__(256) void flash_attn_kernel(
    const __bf16* __restrict__ Qh, const __bf16* __restrict__ Kh,
    const __bf16* __restrict__ VhT, __bf16* __restrict__ AttnOut) {
  __shared__ __bf16 Ps[8][16][40];  // per-wave P bounce (C-layout -> A-frag)

  const int tid  = threadIdx.x;
  const int wv   = tid >> 5;
  const int lane = tid & 31;
  const int hi   = lane >> 4;
  const int ln   = lane & 15;
  const int h    = blockIdx.y;
  const int q0   = blockIdx.x * 128 + wv * 16;   // this wave's 16 query rows

  const __bf16* Qb = Qh + (size_t)h * T_SEQ * HDK;
  const __bf16* Kb = Kh + (size_t)h * T_SEQ * HDK;
  const __bf16* Vb = VhT + (size_t)h * HDV * T_SEQ;

  v16bf qa[2];
  qa[0] = load_frag_a(Qb, HDK, q0 + ln, 0, hi);
  qa[1] = load_frag_a(Qb, HDK, q0 + ln, 32, hi);

  v8f o[4] = {};
  float m[8], l[8];
#pragma unroll
  for (int r = 0; r < 8; ++r) { m[r] = -3.0e38f; l[r] = 0.0f; }

  for (int j0 = 0; j0 < q0 + 16; j0 += 32) {   // causal: keys <= q0+15
    __builtin_prefetch(Kb + (size_t)(j0 + 32) * HDK, 0, 0);  // global_prefetch_b8
    __builtin_prefetch(Vb + (size_t)ln * T_SEQ + j0 + 32, 0, 0);

    // S = Q * K^T / sqrt(64): loads first, then 4 WMMAs
    v16bf kb[2][2];
#pragma unroll
    for (int st = 0; st < 2; ++st)
#pragma unroll
      for (int dk = 0; dk < 2; ++dk)
        kb[st][dk] = load_frag_b(Kb, HDK, j0 + st * 16 + ln, dk * 32, hi);
    v8f s[2] = {};
#pragma unroll
    for (int st = 0; st < 2; ++st)
#pragma unroll
      for (int dk = 0; dk < 2; ++dk)
        s[st] = WMMA_BF16(qa[dk], kb[st][dk], s[st]);

    // Online softmax; C-layout: element (reg r, lane) = row r+8*hi, col 16*st+ln.
    const bool diag = (j0 + 31 > q0);          // block touches the diagonal
#pragma unroll
    for (int r = 0; r < 8; ++r) {
      int qr = q0 + r + 8 * hi;
      float v0 = s[0][r] * 0.125f;             // 1/sqrt(DK)
      float v1 = s[1][r] * 0.125f;
      if (diag) {                              // causal mask only on diagonal blocks
        if (j0 + ln > qr)      v0 = -3.0e38f;
        if (j0 + 16 + ln > qr) v1 = -3.0e38f;
      }
      float rm = fmaxf(v0, v1);
#pragma unroll
      for (int xm = 1; xm < 16; xm <<= 1) rm = fmaxf(rm, __shfl_xor(rm, xm, 32));
      float nm = fmaxf(m[r], rm);
      float p0 = __expf(v0 - nm);
      float p1 = __expf(v1 - nm);
      float rs = p0 + p1;
#pragma unroll
      for (int xm = 1; xm < 16; xm <<= 1) rs += __shfl_xor(rs, xm, 32);
      float corr = __expf(m[r] - nm);
      l[r] = l[r] * corr + rs;
      m[r] = nm;
#pragma unroll
      for (int t = 0; t < 4; ++t) o[t][r] *= corr;
      Ps[wv][r + 8 * hi][ln]      = (__bf16)p0;
      Ps[wv][r + 8 * hi][16 + ln] = (__bf16)p1;
    }
    asm volatile("s_wait_dscnt 0" ::: "memory");  // P visible to whole wave

    // O += P * V  (P: 16x32 A-frag from LDS; V columns contiguous via VhT)
    v16bf pa;
    {
      FragU f;
      const __bf16* p = &Ps[wv][ln][8 * hi];
      f.q[0] = *(const uint4*)(p);
      f.q[1] = *(const uint4*)(p + 16);
      pa = f.v;
    }
    v16bf vb[4];
#pragma unroll
    for (int vt = 0; vt < 4; ++vt)
      vb[vt] = load_frag_b(Vb, T_SEQ, vt * 16 + ln, j0, hi);
#pragma unroll
    for (int vt = 0; vt < 4; ++vt)
      o[vt] = WMMA_BF16(pa, vb[vt], o[vt]);
  }

#pragma unroll
  for (int r = 0; r < 8; ++r) {
    float inv = 1.0f / l[r];
#pragma unroll
    for (int t = 0; t < 4; ++t) o[t][r] *= inv;
  }
#pragma unroll
  for (int t = 0; t < 4; ++t) {
#pragma unroll
    for (int r = 0; r < 8; ++r) {
      int row = q0 + r + 8 * hi;
      int col = h * HDV + t * 16 + ln;
      AttnOut[(size_t)row * (NH * HDV) + col] = (__bf16)o[t][r];
    }
  }
}

// ---- Kernel 3: output projection  Out = AttnOut * w_o^T  (fp32 out) -----
// Block: 256 rows x 64 cols. Wave: 32 rows x 64 cols.
__global__ __launch_bounds__(256) void out_proj_kernel(
    const __bf16* __restrict__ A, const float* __restrict__ Wo,
    float* __restrict__ Out) {
  __shared__ __bf16 As[256][72];
  __shared__ __bf16 Bs[64][72];

  const int tid  = threadIdx.x;
  const int wv   = tid >> 5;
  const int lane = tid & 31;
  const int hi   = lane >> 4;
  const int ln   = lane & 15;
  const int rowBase = blockIdx.x * 256;
  const int colBase = blockIdx.y * 64;

  v8f acc[2][4] = {};

  for (int d0 = 0; d0 < D_MODEL; d0 += 64) {
    __syncthreads();
#pragma unroll
    for (int i = 0; i < 8; ++i) {             // bf16 A tile: async 16B copies if available
      int idx = tid + i * 256;
      int r = idx >> 3, c8 = (idx & 7) * 8;
      copy16_g2l(&As[r][c8], &A[(size_t)(rowBase + r) * D_MODEL + d0 + c8]);
    }
#pragma unroll
    for (int i = 0; i < 4; ++i) {             // fp32 w_o tile -> bf16
      int idx = tid + i * 256;
      int r = idx >> 4, c4 = (idx & 15) * 4;
      float4 wf = *(const float4*)&Wo[(size_t)(colBase + r) * D_MODEL + d0 + c4];
      store4bf(&Bs[r][c4], wf);
    }
    wait_async_done();
    __syncthreads();

    v16bf a[2][2], b[4][2];
#pragma unroll
    for (int rt = 0; rt < 2; ++rt)
#pragma unroll
      for (int dk = 0; dk < 2; ++dk)
        a[rt][dk] = load_frag_a(&As[0][0], 72, wv * 32 + rt * 16 + ln, dk * 32, hi);
#pragma unroll
    for (int t = 0; t < 4; ++t)
#pragma unroll
      for (int dk = 0; dk < 2; ++dk)
        b[t][dk] = load_frag_b(&Bs[0][0], 72, t * 16 + ln, dk * 32, hi);
#pragma unroll
    for (int rt = 0; rt < 2; ++rt)
#pragma unroll
      for (int t = 0; t < 4; ++t)
#pragma unroll
        for (int dk = 0; dk < 2; ++dk)
          acc[rt][t] = WMMA_BF16(a[rt][dk], b[t][dk], acc[rt][t]);
  }

#pragma unroll
  for (int rt = 0; rt < 2; ++rt) {
    const int myRow0 = rowBase + wv * 32 + rt * 16;
#pragma unroll
    for (int t = 0; t < 4; ++t) {
#pragma unroll
      for (int r = 0; r < 8; ++r) {
        int row = myRow0 + r + 8 * hi;
        int col = colBase + t * 16 + ln;
        Out[(size_t)row * D_MODEL + col] = acc[rt][t][r];
      }
    }
  }
}

// ---- Host-side launch ---------------------------------------------------
extern "C" void kernel_launch(void* const* d_in, const int* in_sizes, int n_in,
                              void* d_out, int out_size, void* d_ws, size_t ws_size,
                              hipStream_t stream) {
  (void)in_sizes; (void)n_in; (void)out_size; (void)ws_size;
  const float* q     = (const float*)d_in[0];
  const float* k     = (const float*)d_in[1];
  const float* v     = (const float*)d_in[2];
  // d_in[3] = mask (causal; applied analytically in-kernel)
  const float* freqs = (const float*)d_in[4];
  const float* w_q   = (const float*)d_in[5];
  const float* w_k   = (const float*)d_in[6];
  const float* w_v   = (const float*)d_in[7];
  const float* w_o   = (const float*)d_in[8];
  float* out = (float*)d_out;

  // Workspace layout (bf16): Qh, Kh [H][T][64]; VhT [H][64][T]; AttnOut [T][1024].
  const size_t nPerHead = (size_t)NH * T_SEQ * HDK;   // 4M elements = 8 MB bf16
  __bf16* Qh      = (__bf16*)d_ws;
  __bf16* Kh      = Qh + nPerHead;
  __bf16* VhT     = Kh + nPerHead;
  __bf16* AttnOut = VhT + nPerHead;                   // total 32 MB used

  dim3 gridProj(T_SEQ / 256, NH);
  proj_rope_kernel<<<gridProj, 256, 0, stream>>>(q, w_q, freqs, Qh,  1.0f, 0);
  proj_rope_kernel<<<gridProj, 256, 0, stream>>>(k, w_k, freqs, Kh,  1.0f, 0);
  proj_rope_kernel<<<gridProj, 256, 0, stream>>>(v, w_v, freqs, VhT, 1.0f, 1);

  dim3 gridAttn(T_SEQ / 128, NH);
  flash_attn_kernel<<<gridAttn, 256, 0, stream>>>(Qh, Kh, VhT, AttnOut);

  dim3 gridOut(T_SEQ / 256, D_MODEL / 64);
  out_proj_kernel<<<gridOut, 256, 0, stream>>>(AttnOut, w_o, out);
}
